// ScaledDotProductAttention_17471926960661
// MI455X (gfx1250) — matrix operand order repaired and checked by reading, not compiled, not run on hardware
//
#include <hip/hip_runtime.h>
#include <cstdint>

#define HD   1024
#define LSEQ 2048
#define NB   8
#define BL   (NB * LSEQ)   // 16384 tokens total

typedef __attribute__((ext_vector_type(16))) __bf16 v16bf;
typedef __attribute__((ext_vector_type(8)))  float  v8f;

union FragU {
    v16bf v;
    uint4 q[2];
};

__device__ __forceinline__ unsigned short f2bf(float f) {
    union { float f; unsigned int u; } x; x.f = f;
    unsigned int r = x.u + 0x7FFFu + ((x.u >> 16) & 1u);  // round-to-nearest-even
    return (unsigned short)(r >> 16);
}

__device__ __forceinline__ v8f wmma_bf16(const FragU& a, const FragU& b, v8f c) {
    // D = A(16x32 bf16) * B(32x16 bf16) + C(16x16 f32)
    return __builtin_amdgcn_wmma_f32_16x16x32_bf16(false, a.v, false, b.v,
                                                   (short)0, c, false, false);
}

// A fragment (16x32 bf16): lane row = lane%16; regs 0-3 K = 8*hh..8*hh+7, regs 4-7 K += 16
__device__ __forceinline__ void load_a(FragU& f, const char* base, int row,
                                       int rowStride, int chunkByte, int hh) {
    const char* p = base + row * rowStride + chunkByte + hh * 16;
    f.q[0] = *(const uint4*)(p);
    f.q[1] = *(const uint4*)(p + 32);
}
// B fragment (32x16 bf16) from LDS stored as [n][k]: lane col = lane%16;
// regs 0-3 K = 8*hh..8*hh+7, regs 4-7 K += 16
__device__ __forceinline__ void load_b(FragU& f, const char* base, int row,
                                       int rowStride, int chunkByte, int hh) {
    const char* p = base + row * rowStride + chunkByte + hh * 16;
    f.q[0] = *(const uint4*)(p);
    f.q[1] = *(const uint4*)(p + 32);
}

// ---------------------------------------------------------------------------
// Kernel 1: q/k/v = x @ W^T + b  (bf16 out; q scaled by 1/H; v stored transposed)
// grid: (BL/128, HD/64, 3), block 256 (8 waves as 4m x 2n, each 32x32)
// ---------------------------------------------------------------------------
__global__ __launch_bounds__(256) void qkv_proj(
    const float* __restrict__ x,
    const float* __restrict__ Wq, const float* __restrict__ bq,
    const float* __restrict__ Wk, const float* __restrict__ bk,
    const float* __restrict__ Wv, const float* __restrict__ bv,
    unsigned short* __restrict__ ws)
{
    __shared__ char sm[128 * 80 + 64 * 80];   // A tile 128x32 bf16, B tile 64x32 bf16
    const int AO = 0, BO = 128 * 80;

    const int tid = threadIdx.x, wave = tid >> 5, lane = tid & 31;
    const int ln = lane & 15, hh = lane >> 4;
    const int wm = wave & 3, wn = wave >> 2;
    const int gm0 = blockIdx.x * 128, gn0 = blockIdx.y * 64, z = blockIdx.z;

    const float* W    = (z == 0) ? Wq : (z == 1 ? Wk : Wv);
    const float* bias = (z == 0) ? bq : (z == 1 ? bk : bv);

    v8f acc[2][2];
    const v8f vzero = {0.f,0.f,0.f,0.f,0.f,0.f,0.f,0.f};
    acc[0][0] = vzero; acc[0][1] = vzero; acc[1][0] = vzero; acc[1][1] = vzero;

    for (int kk = 0; kk < HD; kk += 32) {
        // A tile: x[gm0..+128, kk..+32] -> bf16 LDS, row stride 80B
        #pragma unroll
        for (int i = 0; i < 4; i++) {
            int idx = tid + i * 256;
            int row = idx >> 3, c4 = idx & 7;
            float4 xv = *(const float4*)(x + (size_t)(gm0 + row) * HD + kk + c4 * 4);
            unsigned int lo = (unsigned int)f2bf(xv.x) | ((unsigned int)f2bf(xv.y) << 16);
            unsigned int hi = (unsigned int)f2bf(xv.z) | ((unsigned int)f2bf(xv.w) << 16);
            *(uint2*)(sm + AO + row * 80 + c4 * 8) = make_uint2(lo, hi);
        }
        // B tile: W[gn0..+64, kk..+32] (row-major == [n][k] layout)
        #pragma unroll
        for (int i = 0; i < 2; i++) {
            int idx = tid + i * 256;
            int row = idx >> 3, c4 = idx & 7;
            float4 wv = *(const float4*)(W + (size_t)(gn0 + row) * HD + kk + c4 * 4);
            unsigned int lo = (unsigned int)f2bf(wv.x) | ((unsigned int)f2bf(wv.y) << 16);
            unsigned int hi = (unsigned int)f2bf(wv.z) | ((unsigned int)f2bf(wv.w) << 16);
            *(uint2*)(sm + BO + row * 80 + c4 * 8) = make_uint2(lo, hi);
        }
        __syncthreads();

        FragU fa[2], fb[2];
        #pragma unroll
        for (int mi = 0; mi < 2; mi++) load_a(fa[mi], sm + AO, wm * 32 + mi * 16 + ln, 80, 0, hh);
        #pragma unroll
        for (int ni = 0; ni < 2; ni++) load_b(fb[ni], sm + BO, wn * 32 + ni * 16 + ln, 80, 0, hh);
        #pragma unroll
        for (int mi = 0; mi < 2; mi++)
            #pragma unroll
            for (int ni = 0; ni < 2; ni++)
                acc[mi][ni] = wmma_bf16(fa[mi], fb[ni], acc[mi][ni]);
        __syncthreads();
    }

    const float scale = (z == 0) ? (1.0f / (float)HD) : 1.0f;  // fold q/hidden into Q
    const size_t ZOFF = (size_t)BL * HD;
    #pragma unroll
    for (int mi = 0; mi < 2; mi++) {
        #pragma unroll
        for (int ni = 0; ni < 2; ni++) {
            int o = gn0 + wn * 32 + ni * 16 + ln;
            float bv_ = bias[o];
            #pragma unroll
            for (int r = 0; r < 8; r++) {
                int M = gm0 + wm * 32 + mi * 16 + r + 8 * hh;  // token index
                unsigned short hv = f2bf((acc[mi][ni][r] + bv_) * scale);
                if (z < 2) {
                    ws[(size_t)z * ZOFF + (size_t)M * HD + o] = hv;
                } else {
                    int b_ = M >> 11, l_ = M & (LSEQ - 1);     // V stored transposed [B][H][L]
                    ws[2 * ZOFF + (size_t)(b_ * HD + o) * LSEQ + l_] = hv;
                }
            }
        }
    }
}

// ---------------------------------------------------------------------------
// Kernel 2: per-row softmax stats (max, sum) of masked S = Q K^T
// grid: (LSEQ/64, NB), block 256 (8 waves as 4m x 2n of 16x16 S tiles)
// ---------------------------------------------------------------------------
__global__ __launch_bounds__(256) void attn_stats(
    const unsigned short* __restrict__ qws,
    const unsigned short* __restrict__ kws,
    float* __restrict__ smax, float* __restrict__ ssum)
{
    __shared__ char sm[64 * 2064 + 32 * 2064 + 64 * 33 * 4];   // Q | K | S
    const int QO = 0;
    const int KO = 64 * 2064;            // 132096
    const int SO = KO + 32 * 2064;       // 198144

    const int tid = threadIdx.x, wave = tid >> 5, lane = tid & 31;
    const int ln = lane & 15, hh = lane >> 4;
    const int wm = wave & 3, wn = wave >> 2;
    const int m0 = blockIdx.x * 64, b_ = blockIdx.y;
    const int tok0 = b_ * LSEQ + m0;

    // stage Q tile (64 x 1024 bf16) once
    #pragma unroll
    for (int i = 0; i < 32; i++) {
        int idx = tid + i * 256;
        int row = idx >> 7, c4 = idx & 127;
        uint4 v = ((const uint4*)(qws + (size_t)(tok0 + row) * HD))[c4];
        *(uint4*)(sm + QO + row * 2064 + c4 * 16) = v;
    }

    float m_run = -3.402823466e38f, s_run = 0.0f;

    for (int ns = 0; ns < 64; ns++) {
        int n0 = ns * 32;
        __syncthreads();
        // stage K rows n0..n0+32
        #pragma unroll
        for (int i = 0; i < 16; i++) {
            int idx = tid + i * 256;
            int row = idx >> 7, c4 = idx & 127;
            uint4 v = ((const uint4*)(kws + (size_t)(b_ * LSEQ + n0 + row) * HD))[c4];
            *(uint4*)(sm + KO + row * 2064 + c4 * 16) = v;
        }
        __syncthreads();

        v8f s8 = {0.f,0.f,0.f,0.f,0.f,0.f,0.f,0.f};
        for (int c = 0; c < 32; c++) {
            FragU fa, fb;
            load_a(fa, sm + QO, wm * 16 + ln, 2064, c * 64, hh);
            load_b(fb, sm + KO, wn * 16 + ln, 2064, c * 64, hh);
            s8 = wmma_bf16(fa, fb, s8);
        }
        #pragma unroll
        for (int r = 0; r < 8; r++) {
            int M = wm * 16 + r + 8 * hh;
            int N = wn * 16 + ln;
            *(float*)(sm + SO + (M * 33 + N) * 4) = s8[r];
        }
        __syncthreads();

        if (tid < 64) {  // thread-per-row online max/sum
            int m_l = m0 + tid;
            float sv[32];
            #pragma unroll
            for (int j = 0; j < 32; j++) {
                float s = *(const float*)(sm + SO + (tid * 33 + j) * 4);
                sv[j] = (n0 + j == m_l) ? -3.402823466e38f : s;   // mask diagonal
            }
            float tm = m_run;
            #pragma unroll
            for (int j = 0; j < 32; j++) tm = fmaxf(tm, sv[j]);
            float acc = s_run * __expf(m_run - tm);
            #pragma unroll
            for (int j = 0; j < 32; j++) acc += __expf(sv[j] - tm);
            m_run = tm; s_run = acc;
        }
    }
    if (tid < 64) {
        smax[tok0 + tid] = m_run;
        ssum[tok0 + tid] = s_run;
    }
}

// ---------------------------------------------------------------------------
// Kernel 3: out = softmax(S) @ V, recomputing S, h-chunked (512 cols per block)
// grid: (LSEQ/64, HD/512, NB), block 256
//   S phase : waves 4m x 2n (16x16 tiles over 64x32)
//   PV phase: waves 4m x 2h (each 16 rows x 256 h cols = 16 wmma tiles)
// ---------------------------------------------------------------------------
__global__ __launch_bounds__(256) void attn_pv(
    const unsigned short* __restrict__ qws,
    const unsigned short* __restrict__ kws,
    const unsigned short* __restrict__ vtws,   // V^T [B][H][L]
    const float* __restrict__ smax, const float* __restrict__ ssum,
    float* __restrict__ out)
{
    __shared__ char sm[64 * 2064 + 32 * 2064 + 512 * 80 + 64 * 80 + 512]; // Q|K|Vt|P|stats
    const int QO  = 0;
    const int KO  = 64 * 2064;           // 132096
    const int VO  = KO + 32 * 2064;      // 198144
    const int PO  = VO + 512 * 80;       // 239104
    const int STO = PO + 64 * 80;        // 244224 (+512)

    const int tid = threadIdx.x, wave = tid >> 5, lane = tid & 31;
    const int ln = lane & 15, hh = lane >> 4;
    const int wm = wave & 3, wn = wave >> 2;   // wn doubles as h-half in PV phase
    const int m0 = blockIdx.x * 64;
    const int h0 = blockIdx.y * 512;
    const int b_ = blockIdx.z;
    const int tok0 = b_ * LSEQ + m0;

    #pragma unroll
    for (int i = 0; i < 32; i++) {
        int idx = tid + i * 256;
        int row = idx >> 7, c4 = idx & 127;
        uint4 v = ((const uint4*)(qws + (size_t)(tok0 + row) * HD))[c4];
        *(uint4*)(sm + QO + row * 2064 + c4 * 16) = v;
    }
    if (tid < 64) {
        *(float*)(sm + STO + tid * 4)       = smax[tok0 + tid];
        *(float*)(sm + STO + 256 + tid * 4) = 1.0f / ssum[tok0 + tid];
    }

    v8f acc[16];
    const v8f vzero = {0.f,0.f,0.f,0.f,0.f,0.f,0.f,0.f};
    #pragma unroll
    for (int t = 0; t < 16; t++) acc[t] = vzero;

    for (int ns = 0; ns < 64; ns++) {
        int n0 = ns * 32;
        __syncthreads();
        #pragma unroll
        for (int i = 0; i < 16; i++) {     // K rows n0..n0+32 (full H)
            int idx = tid + i * 256;
            int row = idx >> 7, c4 = idx & 127;
            uint4 v = ((const uint4*)(kws + (size_t)(b_ * LSEQ + n0 + row) * HD))[c4];
            *(uint4*)(sm + KO + row * 2064 + c4 * 16) = v;
        }
        #pragma unroll
        for (int i = 0; i < 8; i++) {      // V^T rows h0..h0+512, cols n0..n0+32
            int idx = tid + i * 256;
            int hr = idx >> 2, c4 = idx & 3;
            uint4 v = ((const uint4*)(vtws + (size_t)(b_ * HD + h0 + hr) * LSEQ + n0))[c4];
            *(uint4*)(sm + VO + hr * 80 + c4 * 16) = v;
        }
        __syncthreads();

        // S = Q K^T for this 64x32 slab
        v8f s8 = {0.f,0.f,0.f,0.f,0.f,0.f,0.f,0.f};
        for (int c = 0; c < 32; c++) {
            FragU fa, fb;
            load_a(fa, sm + QO, wm * 16 + ln, 2064, c * 64, hh);
            load_b(fb, sm + KO, wn * 16 + ln, 2064, c * 64, hh);
            s8 = wmma_bf16(fa, fb, s8);
        }
        // P = exp(S - max) / sum  (bf16), diagonal masked to 0
        #pragma unroll
        for (int r = 0; r < 8; r++) {
            int M = wm * 16 + r + 8 * hh;
            int n_l = n0 + wn * 16 + ln;
            float p;
            if (m0 + M == n_l) {
                p = 0.0f;
            } else {
                float mx = *(const float*)(sm + STO + M * 4);
                float is = *(const float*)(sm + STO + 256 + M * 4);
                p = __expf(s8[r] - mx) * is;
            }
            *(unsigned short*)(sm + PO + M * 80 + (wn * 16 + ln) * 2) = f2bf(p);
        }
        __syncthreads();

        // PV: acc += P(16x32) * V(32x16) per h tile
        FragU fp;
        load_a(fp, sm + PO, wm * 16 + ln, 80, 0, hh);
        #pragma unroll
        for (int ti = 0; ti < 16; ti++) {
            FragU fv;
            load_b(fv, sm + VO, wn * 256 + ti * 16 + ln, 80, 0, hh);
            acc[ti] = wmma_bf16(fp, fv, acc[ti]);
        }
    }

    #pragma unroll
    for (int ti = 0; ti < 16; ti++) {
        int o = h0 + wn * 256 + ti * 16 + ln;
        #pragma unroll
        for (int r = 0; r < 8; r++) {
            int M = wm * 16 + r + 8 * hh;
            out[(size_t)(tok0 + M) * HD + o] = acc[ti][r];
        }
    }
}

// ---------------------------------------------------------------------------
extern "C" void kernel_launch(void* const* d_in, const int* in_sizes, int n_in,
                              void* d_out, int out_size, void* d_ws, size_t ws_size,
                              hipStream_t stream)
{
    const float* x  = (const float*)d_in[0];
    const float* Wq = (const float*)d_in[1];
    const float* bq = (const float*)d_in[2];
    const float* Wk = (const float*)d_in[3];
    const float* bk = (const float*)d_in[4];
    const float* Wv = (const float*)d_in[5];
    const float* bv = (const float*)d_in[6];

    unsigned short* ws = (unsigned short*)d_ws;
    const size_t ZOFF = (size_t)BL * HD;                       // elements per q/k/v buffer
    float* smax = (float*)((char*)d_ws + 3 * ZOFF * sizeof(unsigned short));
    float* ssum = smax + BL;
    float* out  = (float*)d_out;

    qkv_proj  <<<dim3(BL / 128, HD / 64, 3), 256, 0, stream>>>(x, Wq, bq, Wk, bk, Wv, bv, ws);
    attn_stats<<<dim3(LSEQ / 64, NB),        256, 0, stream>>>(ws, ws + ZOFF, smax, ssum);
    attn_pv   <<<dim3(LSEQ / 64, HD / 512, NB), 256, 0, stream>>>(ws, ws + ZOFF, ws + 2 * ZOFF,
                                                                  smax, ssum, out);
}
